// SepConv_81466939671249
// MI455X (gfx1250) — matrix operand order repaired and checked by reading, not compiled, not run on hardware
//
#include <hip/hip_runtime.h>

typedef _Float16 f16;
typedef __attribute__((ext_vector_type(16))) _Float16 v16h;
typedef __attribute__((ext_vector_type(8)))  _Float16 v8h;
typedef __attribute__((ext_vector_type(8)))  float    v8f;

#define BATCH   16
#define HH      224
#define WW      224
#define CC      96
#define TILE_PX 128          // pixels per workgroup (8 waves x 16 M-rows)
#define LDA     104          // padded LDS row stride in f16 elements (breaks bank conflicts)

__launch_bounds__(256)
__global__ void sepconv_fused(const float* __restrict__ x,
                              const float* __restrict__ dwk,   // (3,3,1,96)
                              const float* __restrict__ dwb,   // (96)
                              const float* __restrict__ pwk,   // (1,1,96,96) [ci][co]
                              const float* __restrict__ pwb,   // (96)
                              const float* __restrict__ gamma,
                              const float* __restrict__ beta,
                              const float* __restrict__ mmean,
                              const float* __restrict__ mvar,
                              float* __restrict__ out)
{
    __shared__ __align__(16) f16  sA[TILE_PX * LDA]; // depthwise output tile (M x K), fp16
    __shared__ __align__(16) f16  sB[CC * LDA];      // pointwise weights, N-major (N x K), fp16
    __shared__ float sW[9 * CC];                     // depthwise 3x3 weights
    __shared__ float sBias[CC];                      // depthwise bias
    __shared__ float sScale[CC];                     // BN scale
    __shared__ float sShift[CC];                     // BN shift (pw_bias folded in)

    const int tid = threadIdx.x;
    const int w0  = blockIdx.x * TILE_PX;
    const int h   = blockIdx.y;
    const int b   = blockIdx.z;

    // ---- prefetch the three input rows for this tile toward L2 ----
    {
        const int span = ((WW - w0) < TILE_PX ? (WW - w0) : TILE_PX) * CC; // floats
        #pragma unroll
        for (int dh = 0; dh < 3; ++dh) {
            int hh = h - 1 + dh;
            if ((unsigned)hh < (unsigned)HH) {
                const float* rowp = x + (((size_t)b * HH + hh) * WW + w0) * CC;
                for (int off = tid * 64; off < span; off += 256 * 64)
                    __builtin_prefetch(rowp + off, 0, 0);
            }
        }
    }

    // ---- stage constants into LDS ----
    for (int i = tid; i < 9 * CC; i += 256) sW[i] = dwk[i];
    if (tid < CC) {
        sBias[tid] = dwb[tid];
        float inv  = gamma[tid] * rsqrtf(mvar[tid] + 1e-3f);
        sScale[tid] = inv;
        sShift[tid] = beta[tid] + (pwb[tid] - mmean[tid]) * inv;
    }
    // pwk[ci*CC + co] -> sB[co*LDA + ci]  (transpose to N-major for contiguous-K fragments)
    for (int i = tid; i < CC * CC; i += 256) {
        int ci = i / CC, co = i % CC;
        sB[co * LDA + ci] = (f16)pwk[i];
    }
    __syncthreads();

    // ---- stage 1: ReLU -> depthwise 3x3 (+bias) into sA as fp16 ----
    for (int i = tid; i < TILE_PX * CC; i += 256) {
        int px = i / CC, c = i % CC;
        int w  = w0 + px;
        float acc = 0.0f;
        if (w < WW) {
            acc = sBias[c];
            #pragma unroll
            for (int dh = -1; dh <= 1; ++dh) {
                int hh = h + dh;
                if ((unsigned)hh < (unsigned)HH) {
                    const float* xr = x + (((size_t)b * HH + hh) * WW) * CC + c;
                    #pragma unroll
                    for (int dw = -1; dw <= 1; ++dw) {
                        int ww = w + dw;
                        if ((unsigned)ww < (unsigned)WW) {
                            float xv = xr[(size_t)ww * CC];
                            xv = fmaxf(xv, 0.0f);                    // leading ReLU
                            acc = fmaf(xv, sW[((dh + 1) * 3 + (dw + 1)) * CC + c], acc);
                        }
                    }
                }
            }
        }
        sA[px * LDA + c] = (f16)acc;
    }
    __syncthreads();

    // ---- stage 2: pointwise 1x1 as WMMA GEMM (M=128 tile, N=96, K=96) ----
    const int lane  = tid & 31;
    const int wave  = tid >> 5;            // 0..7, each wave owns 16 M-rows
    const int ml    = lane & 15;           // row/col within fragment
    const int hiLn  = lane >> 4;           // 0 for lanes 0-15, 1 for 16-31

    const f16* arow = &sA[(wave * 16 + ml) * LDA];

    v8f acc[6];
    const v8f vzero = {0.f, 0.f, 0.f, 0.f, 0.f, 0.f, 0.f, 0.f};
    #pragma unroll
    for (int n = 0; n < 6; ++n) acc[n] = vzero;

    #pragma unroll
    for (int kk = 0; kk < 3; ++kk) {
        const int k0 = kk * 32;
        // A fragment (16x32 f16): low lanes hold K 0..7 & 16..23, high lanes 8..15 & 24..31
        union { v16h v; v8h p[2]; } af;
        af.p[0] = *(const v8h*)(arow + k0 + hiLn * 8);
        af.p[1] = *(const v8h*)(arow + k0 + 16 + hiLn * 8);
        #pragma unroll
        for (int n = 0; n < 6; ++n) {
            // B fragment (32x16 f16): column N=lane&15; low lanes K 0..7 & 16..23, high lanes 8..15 & 24..31
            const f16* brow = &sB[(n * 16 + ml) * LDA];
            union { v16h v; v8h p[2]; } bf;
            bf.p[0] = *(const v8h*)(brow + k0 + hiLn * 8);
            bf.p[1] = *(const v8h*)(brow + k0 + 16 + hiLn * 8);
            acc[n] = __builtin_amdgcn_wmma_f32_16x16x32_f16(
                false, af.v, false, bf.v, (short)0, acc[n], false, false);
        }
    }

    // ---- epilogue: BN scale/shift, store fp32 (C/D layout: VGPR r -> M=r(+8), N=lane&15) ----
    #pragma unroll
    for (int n = 0; n < 6; ++n) {
        const int cO = n * 16 + ml;
        const float sc = sScale[cO];
        const float sh = sShift[cO];
        #pragma unroll
        for (int r = 0; r < 8; ++r) {
            int m = wave * 16 + r + (hiLn << 3);
            int w = w0 + m;
            if (w < WW) {
                out[(((size_t)b * HH + h) * WW + w) * CC + cO] = acc[n][r] * sc + sh;
            }
        }
    }
}

extern "C" void kernel_launch(void* const* d_in, const int* in_sizes, int n_in,
                              void* d_out, int out_size, void* d_ws, size_t ws_size,
                              hipStream_t stream) {
    const float* x     = (const float*)d_in[0];
    const float* dwk   = (const float*)d_in[1];
    const float* dwb   = (const float*)d_in[2];
    const float* pwk   = (const float*)d_in[3];
    const float* pwb   = (const float*)d_in[4];
    const float* gamma = (const float*)d_in[5];
    const float* beta  = (const float*)d_in[6];
    const float* mmean = (const float*)d_in[7];
    const float* mvar  = (const float*)d_in[8];
    float* out = (float*)d_out;

    dim3 grid((WW + TILE_PX - 1) / TILE_PX, HH, BATCH);  // (2, 224, 16)
    sepconv_fused<<<grid, 256, 0, stream>>>(x, dwk, dwb, pwk, pwb,
                                            gamma, beta, mmean, mvar, out);
}
